// SpatialAttention_13434657702215
// MI455X (gfx1250) — matrix operand order repaired and checked by reading, not compiled, not run on hardware
//
#include <hip/hip_runtime.h>
#include <hip/hip_bf16.h>
#include <math.h>

typedef __attribute__((ext_vector_type(16))) _Float16 v16h;
typedef __attribute__((ext_vector_type(8)))  _Float16 v8h;
typedef __attribute__((ext_vector_type(8)))  float    v8f;
typedef __attribute__((ext_vector_type(4)))  int      v4i;

#define BB 8
#define HH 32
#define WW 32
#define CC 256
#define LL 1024
#define KP 2304   // 9*C
#define C2 512    // 2*C

#if defined(__has_builtin)
#if __has_builtin(__builtin_amdgcn_global_load_async_to_lds_b128) && \
    __has_builtin(__builtin_amdgcn_s_wait_asynccnt)
#define ASYNC_LDS 1
#endif
#endif

// 16-byte global -> LDS copy; async (ASYNCcnt) when available on this toolchain.
__device__ __forceinline__ void cp16(_Float16* lds, const _Float16* g) {
#ifdef ASYNC_LDS
  __builtin_amdgcn_global_load_async_to_lds_b128(
      (__attribute__((address_space(1))) v4i*)g,
      (__attribute__((address_space(3))) v4i*)lds, 0, 0);
#else
  *(v8h*)lds = *(const v8h*)g;
#endif
}
__device__ __forceinline__ void cp_wait() {
#ifdef ASYNC_LDS
  __builtin_amdgcn_s_wait_asynccnt(0);
#endif
}

// ---------------- WMMA fragment helpers (16x16x32 f16, wave32) ----------------
// A (16x32) row-major LDS, ld = 40 halves (80B rows: 16B aligned, conflict-free).
// lanes 0-15: row M=lane, K in {0..15}; lanes 16-31: row M=lane-16, K in {16..31}.
__device__ __forceinline__ v16h load_frag_a(const _Float16* base, int row, int lane) {
  int m = row + (lane & 15);
  int kb = (lane < 16) ? 0 : 16;
  const _Float16* r = base + m * 40 + kb;
  v8h lo = *(const v8h*)r;
  v8h hi = *(const v8h*)(r + 8);
  return __builtin_shufflevector(lo, hi, 0,1,2,3,4,5,6,7,8,9,10,11,12,13,14,15);
}
// B staged in fragment order: Bf[ntile][lane][16 halves] contiguous per lane.
__device__ __forceinline__ v16h load_frag_bf(const _Float16* bf, int ntile, int lane) {
  const _Float16* p = bf + (size_t)(ntile * 32 + lane) * 16;
  v8h lo = *(const v8h*)p;
  v8h hi = *(const v8h*)(p + 8);
  return __builtin_shufflevector(lo, hi, 0,1,2,3,4,5,6,7,8,9,10,11,12,13,14,15);
}
// Fragment-order index for element (k, n) of a 32xN B tile.
__device__ __forceinline__ int bf_idx(int kk, int n) {
  return ((n >> 4) * 32 + (n & 15) + ((kk >= 16) ? 16 : 0)) * 16 + (kk & 15);
}

// ---------------- prep: im2col patch matrix P[b][q][st*256+c] (f16) ----------------
__global__ void prep_patches(const float* __restrict__ x, _Float16* __restrict__ P) {
  size_t idx = (size_t)blockIdx.x * blockDim.x + threadIdx.x;
  if (idx >= (size_t)BB * LL * KP) return;
  int c = (int)(idx % CC);
  size_t r = idx / CC;
  int st = (int)(r % 9);
  size_t r2 = r / 9;
  int q = (int)(r2 % LL);
  int b = (int)(r2 / LL);
  int s = st / 3, t = st % 3;
  int qi = q >> 5, qj = q & 31;
  int yy = qi + s - 1, xx = qj + t - 1;
  float v = 0.f;
  if (yy >= 0 && yy < HH && xx >= 0 && xx < WW)
    v = x[(((size_t)b * HH + yy) * WW + xx) * CC + c];
  P[idx] = (_Float16)v;
}

// ---------------- prep: per-patch inverse norm + mask-mean validity ----------------
__global__ void prep_scl_mm(const float* __restrict__ x, const float* __restrict__ mask,
                            float* __restrict__ scl, float* __restrict__ mm) {
  int b = blockIdx.x >> 10, q = blockIdx.x & 1023;
  int qi = q >> 5, qj = q & 31, tid = threadIdx.x;
  float s = 0.f;
#pragma unroll
  for (int st = 0; st < 9; ++st) {
    int yy = qi + st / 3 - 1, xx = qj + st % 3 - 1;
    if (yy >= 0 && yy < HH && xx >= 0 && xx < WW) {
      float v = x[(((size_t)b * HH + yy) * WW + xx) * CC + tid];
      s += v * v;
    }
  }
  __shared__ float red[256];
  red[tid] = s; __syncthreads();
  for (int w = 128; w > 0; w >>= 1) { if (tid < w) red[tid] += red[tid + w]; __syncthreads(); }
  if (tid == 0) {
    scl[blockIdx.x] = 1.f / fmaxf(sqrtf(red[0]), 1e-4f);
    float msum = 0.f;
#pragma unroll
    for (int st = 0; st < 9; ++st) {
      int yy = qi + st / 3 - 1, xx = qj + st % 3 - 1;
      if (yy >= 0 && yy < HH && xx >= 0 && xx < WW)
        msum += mask[((size_t)b * HH + yy) * WW + xx];
    }
    mm[blockIdx.x] = ((msum * (1.f / 9.f)) == 1.f) ? 1.f : 0.f;
  }
}

// ---------------- prep: f32 -> f16 weight conversion ----------------
__global__ void prep_w(const float* __restrict__ cw, const float* __restrict__ c1w,
                       const float* __restrict__ c2w, _Float16* __restrict__ hcw,
                       _Float16* __restrict__ hc1w, _Float16* __restrict__ hc2w) {
  int i = blockIdx.x * 256 + threadIdx.x;
  if (i < 9 * C2 * CC) { hcw[i] = (_Float16)cw[i]; hc2w[i] = (_Float16)c2w[i]; }
  if (i < C2 * CC) hc1w[i] = (_Float16)c1w[i];
}

// ---------------- prep: ycat high half = x (f16) ----------------
__global__ void prep_xcat(const float* __restrict__ x, _Float16* __restrict__ ycat) {
  size_t idx = (size_t)blockIdx.x * blockDim.x + threadIdx.x;
  if (idx >= (size_t)BB * LL * CC) return;
  int c = (int)(idx % CC);
  size_t p = idx / CC;           // b*1024 + p
  ycat[p * C2 + CC + c] = (_Float16)x[idx];
}

// ---------------- GEMM1: S[p][q] = (P Pt)[p][q] * scl[q]  (64x128 tiles) ----------------
__global__ void gemm_scores(const _Float16* __restrict__ P, const float* __restrict__ scl,
                            float* __restrict__ S) {
  int b = blockIdx.x, pt = blockIdx.y, qt = blockIdx.z;
  const _Float16* Pb = P + (size_t)b * LL * KP;
  __shared__ __align__(16) _Float16 As[64 * 40];
  __shared__ __align__(16) _Float16 Bf[8 * 32 * 16];
  int tid = threadIdx.x, lane = tid & 31, wave = tid >> 5;
  int wm = (wave >> 1) * 16;
  int nt0 = (wave & 1) * 4;
  v8f acc[4] = {};
  int p0 = pt * 64, q0 = qt * 128;
  for (int k0 = 0; k0 < KP; k0 += 32) {
    { // A: 64 rows x 32 k, K-contiguous chunks of 8 halves
      int r = tid >> 2, c = tid & 3;
      cp16(As + r * 40 + 8 * c, Pb + (size_t)(p0 + r) * KP + k0 + 8 * c);
    }
    // B: 128 n x 32 k, K-contiguous source -> fragment-order contiguous chunks
    for (int e = tid; e < 512; e += 256) {
      int n = e >> 2, c = e & 3;
      int ln = (n & 15) + ((c >= 2) ? 16 : 0);
      cp16(Bf + ((size_t)(n >> 4) * 32 + ln) * 16 + 8 * (c & 1),
           Pb + (size_t)(q0 + n) * KP + k0 + 8 * c);
    }
    cp_wait();
    __syncthreads();
    v16h a = load_frag_a(As, wm, lane);
#pragma unroll
    for (int t = 0; t < 4; ++t) {
      v16h bfr = load_frag_bf(Bf, nt0 + t, lane);
      acc[t] = __builtin_amdgcn_wmma_f32_16x16x32_f16(false, a, false, bfr, (short)0, acc[t], false, false);
    }
    __syncthreads();
  }
  float* Sb = S + (size_t)b * LL * LL;
  const float* sclb = scl + b * LL;
  int nlo = lane & 15, mo = (lane < 16) ? 0 : 8;
#pragma unroll
  for (int t = 0; t < 4; ++t) {
    int col = q0 + (nt0 + t) * 16 + nlo;
    float sc = sclb[col];
#pragma unroll
    for (int r = 0; r < 8; ++r) {
      int m = p0 + wm + mo + r;
      Sb[(size_t)m * LL + col] = acc[t][r] * sc;
    }
  }
}

// ---------------- fuse (diag propagate x2 with perm) + masked softmax ----------------
__device__ __forceinline__ float F1v(const float* __restrict__ Sb, int r, int c) {
  float v = Sb[(size_t)r * LL + c];
  if (r > 0 && c > 0)       v += Sb[(size_t)(r - 1) * LL + (c - 1)];
  if (r < LL-1 && c < LL-1) v += Sb[(size_t)(r + 1) * LL + (c + 1)];
  return v;
}
__device__ __forceinline__ int permi(int x) { return ((x & 31) << 5) | (x >> 5); }

__global__ void fuse_softmax(const float* __restrict__ S, const float* __restrict__ mm,
                             _Float16* __restrict__ A) {
  int b = blockIdx.x >> 10, p = blockIdx.x & 1023;
  const float* Sb = S + (size_t)b * LL * LL;
  const float* mmb = mm + b * LL;
  int tid = threadIdx.x;
  int pm = permi(p);
  int rA = (pm > 0)      ? permi(pm - 1) : -1;
  int rB = (pm < LL - 1) ? permi(pm + 1) : -1;
  __shared__ float red[256];
  float vals[4];
  float lmax = -1e30f;
#pragma unroll
  for (int m = 0; m < 4; ++m) {
    int q = tid + (m << 8);
    float f = F1v(Sb, p, q);
    int qm = permi(q);
    if (rA >= 0 && qm > 0)      f += F1v(Sb, rA, permi(qm - 1));
    if (rB >= 0 && qm < LL - 1) f += F1v(Sb, rB, permi(qm + 1));
    float v = f * mmb[q] * 10.f;  // SOFTMAX_SCALE
    vals[m] = v;
    lmax = fmaxf(lmax, v);
  }
  red[tid] = lmax; __syncthreads();
  for (int w = 128; w > 0; w >>= 1) { if (tid < w) red[tid] = fmaxf(red[tid], red[tid + w]); __syncthreads(); }
  float mx = red[0]; __syncthreads();
  float lsum = 0.f;
#pragma unroll
  for (int m = 0; m < 4; ++m) { vals[m] = __expf(vals[m] - mx); lsum += vals[m]; }
  red[tid] = lsum; __syncthreads();
  for (int w = 128; w > 0; w >>= 1) { if (tid < w) red[tid] += red[tid + w]; __syncthreads(); }
  float inv = 1.f / red[0];
  _Float16* Ab = A + ((size_t)b * LL + p) * LL;
#pragma unroll
  for (int m = 0; m < 4; ++m) {
    int q = tid + (m << 8);
    Ab[q] = (_Float16)(vals[m] * inv * mmb[q]);
  }
}

// ---------------- GEMM2: deconv y[p][c] = sum_{s,t,q} A[shift(p;s,t)][q]*P[q][st*256+c]
__global__ void gemm_deconv(const _Float16* __restrict__ A, const _Float16* __restrict__ P,
                            _Float16* __restrict__ ycat) {
  int b = blockIdx.x, pt = blockIdx.y, ct = blockIdx.z;
  const _Float16* Ab = A + (size_t)b * LL * LL;
  const _Float16* Pb = P + (size_t)b * LL * KP;
  __shared__ __align__(16) _Float16 As[64 * 40];
  __shared__ __align__(16) _Float16 Bf[8 * 32 * 16];
  int tid = threadIdx.x, lane = tid & 31, wave = tid >> 5;
  int wm = (wave >> 1) * 16;
  int nt0 = (wave & 1) * 4;
  v8f acc[4] = {};
  int p0 = pt * 64, c0 = ct * 128;
  for (int st = 0; st < 9; ++st) {
    int s = st / 3, t3 = st % 3;
    for (int q0 = 0; q0 < LL; q0 += 32) {
      { // A: shifted attention rows (zero-filled at borders), 16B chunks
        int r = tid >> 2, c = tid & 3;
        int p = p0 + r, i = p >> 5, j = p & 31;
        int ii = i + 1 - s, jj = j + 1 - t3;
        v8h val = {};
        if (ii >= 0 && ii < HH && jj >= 0 && jj < WW)
          val = *(const v8h*)(Ab + (size_t)(ii * WW + jj) * LL + q0 + 8 * c);
        *(v8h*)(As + r * 40 + 8 * c) = val;
      }
      // B: N-contiguous source -> coalesced v8h load + fragment-order scatter
      for (int e = tid; e < 512; e += 256) {
        int kk = e >> 4, d = e & 15;
        v8h v = *(const v8h*)(Pb + (size_t)(q0 + kk) * KP + st * CC + c0 + 8 * d);
#pragma unroll
        for (int i2 = 0; i2 < 8; ++i2) Bf[bf_idx(kk, 8 * d + i2)] = v[i2];
      }
      __syncthreads();
      v16h a = load_frag_a(As, wm, lane);
#pragma unroll
      for (int t = 0; t < 4; ++t) {
        v16h bfr = load_frag_bf(Bf, nt0 + t, lane);
        acc[t] = __builtin_amdgcn_wmma_f32_16x16x32_f16(false, a, false, bfr, (short)0, acc[t], false, false);
      }
      __syncthreads();
    }
  }
  _Float16* Yb = ycat + (size_t)b * LL * C2;
  int nlo = lane & 15, mo = (lane < 16) ? 0 : 8;
#pragma unroll
  for (int t = 0; t < 4; ++t) {
    int col = c0 + (nt0 + t) * 16 + nlo;
#pragma unroll
    for (int r = 0; r < 8; ++r) {
      int p = p0 + wm + mo + r;
      Yb[(size_t)p * C2 + col] = (_Float16)acc[t][r];
    }
  }
}

// ---------------- GEMM3: trailing convs + sigmoid gate epilogue ----------------
__global__ void gemm_out(const _Float16* __restrict__ ycat, const _Float16* __restrict__ cw,
                         const _Float16* __restrict__ c1w, const _Float16* __restrict__ c2w,
                         const float* __restrict__ cb, const float* __restrict__ c1b,
                         const float* __restrict__ c2b, float* __restrict__ out) {
  int b = blockIdx.x, pt = blockIdx.y, ct = blockIdx.z;
  const _Float16* Yb = ycat + (size_t)b * LL * C2;
  __shared__ __align__(16) _Float16 As[64 * 40];
  __shared__ __align__(16) _Float16 Bf0[8 * 32 * 16];
  __shared__ __align__(16) _Float16 Bf2[8 * 32 * 16];
  __shared__ __align__(16) _Float16 Bf1[8 * 32 * 16];
  int tid = threadIdx.x, lane = tid & 31, wave = tid >> 5;
  int wm = (wave >> 1) * 16;
  int nt0 = (wave & 1) * 4;
  v8f ao[4] = {}, a1[4] = {}, a2[4] = {};
  int p0 = pt * 64, c0 = ct * 128;
  for (int uv = 0; uv < 9; ++uv) {
    int u = uv / 3, v = uv % 3;
    for (int k0 = 0; k0 < C2; k0 += 32) {
      { // A: shifted ycat rows (zero-filled at borders)
        int r = tid >> 2, c = tid & 3;
        int p = p0 + r, i = p >> 5, j = p & 31;
        int ii = i + u - 1, jj = j + v - 1;
        v8h val = {};
        if (ii >= 0 && ii < HH && jj >= 0 && jj < WW)
          val = *(const v8h*)(Yb + (size_t)(ii * WW + jj) * C2 + k0 + 8 * c);
        *(v8h*)(As + r * 40 + 8 * c) = val;
      }
      // weights: N-contiguous -> fragment-order scatter
      for (int e = tid; e < 512; e += 256) {
        int kk = e >> 4, d = e & 15;
        size_t wrow = (size_t)(uv * C2 + k0 + kk) * CC + c0 + 8 * d;
        v8h w0 = *(const v8h*)(cw + wrow);
        v8h w2 = *(const v8h*)(c2w + wrow);
#pragma unroll
        for (int i2 = 0; i2 < 8; ++i2) {
          int bi = bf_idx(kk, 8 * d + i2);
          Bf0[bi] = w0[i2];
          Bf2[bi] = w2[i2];
        }
        if (uv == 4) {
          v8h w1 = *(const v8h*)(c1w + (size_t)(k0 + kk) * CC + c0 + 8 * d);
#pragma unroll
          for (int i2 = 0; i2 < 8; ++i2) Bf1[bf_idx(kk, 8 * d + i2)] = w1[i2];
        }
      }
      __syncthreads();
      v16h a = load_frag_a(As, wm, lane);
#pragma unroll
      for (int t = 0; t < 4; ++t) {
        v16h b0 = load_frag_bf(Bf0, nt0 + t, lane);
        ao[t] = __builtin_amdgcn_wmma_f32_16x16x32_f16(false, a, false, b0, (short)0, ao[t], false, false);
        v16h b2 = load_frag_bf(Bf2, nt0 + t, lane);
        a2[t] = __builtin_amdgcn_wmma_f32_16x16x32_f16(false, a, false, b2, (short)0, a2[t], false, false);
      }
      if (uv == 4) {
#pragma unroll
        for (int t = 0; t < 4; ++t) {
          v16h b1 = load_frag_bf(Bf1, nt0 + t, lane);
          a1[t] = __builtin_amdgcn_wmma_f32_16x16x32_f16(false, a, false, b1, (short)0, a1[t], false, false);
        }
      }
      __syncthreads();
    }
  }
  int nlo = lane & 15, mo = (lane < 16) ? 0 : 8;
#pragma unroll
  for (int t = 0; t < 4; ++t) {
    int col = c0 + (nt0 + t) * 16 + nlo;
    float bc = cb[col], b1v = c1b[col], b2v = c2b[col];
#pragma unroll
    for (int r = 0; r < 8; ++r) {
      int p = p0 + wm + mo + r;
      float cc = ao[t][r] + bc;
      float s1 = 1.f / (1.f + __expf(-(a1[t][r] + b1v)));
      float s2 = 1.f / (1.f + __expf(-(a2[t][r] + b2v)));
      out[((size_t)b * LL + p) * CC + col] = 0.5f * (s1 + s2) * cc;
    }
  }
}

__global__ void fill_ones(float* __restrict__ out) {
  int i = blockIdx.x * 256 + threadIdx.x;
  if (i < BB * LL) out[i] = 1.f;
}

extern "C" void kernel_launch(void* const* d_in, const int* in_sizes, int n_in,
                              void* d_out, int out_size, void* d_ws, size_t ws_size,
                              hipStream_t stream) {
  const float* x     = (const float*)d_in[0];
  const float* mask  = (const float*)d_in[1];
  const float* cw    = (const float*)d_in[2];
  const float* cb    = (const float*)d_in[3];
  const float* c1w   = (const float*)d_in[4];
  const float* c1b   = (const float*)d_in[5];
  const float* c2w   = (const float*)d_in[6];
  const float* c2b   = (const float*)d_in[7];
  float* out = (float*)d_out;

  char* ws = (char*)d_ws;
  size_t offP   = 0;                                      // f16 [B][L][2304]
  size_t offS   = offP   + (size_t)BB * LL * KP * 2;      // f32 [B][L][L]
  size_t offA   = offS   + (size_t)BB * LL * LL * 4;      // f16 [B][L][L]
  size_t offY   = offA   + (size_t)BB * LL * LL * 2;      // f16 [B][L][512]
  size_t offScl = offY   + (size_t)BB * LL * C2 * 2;      // f32 [B][L]
  size_t offMm  = offScl + (size_t)BB * LL * 4;           // f32 [B][L]
  size_t offCw  = offMm  + (size_t)BB * LL * 4;           // f16 [9*512*256]
  size_t offC1  = offCw  + (size_t)9 * C2 * CC * 2;       // f16 [512*256]
  size_t offC2w = offC1  + (size_t)C2 * CC * 2;           // f16 [9*512*256]

  _Float16* P    = (_Float16*)(ws + offP);
  float*    S    = (float*)(ws + offS);
  _Float16* Amat = (_Float16*)(ws + offA);
  _Float16* ycat = (_Float16*)(ws + offY);
  float*    scl  = (float*)(ws + offScl);
  float*    mm   = (float*)(ws + offMm);
  _Float16* hcw  = (_Float16*)(ws + offCw);
  _Float16* hc1w = (_Float16*)(ws + offC1);
  _Float16* hc2w = (_Float16*)(ws + offC2w);

  // prep
  {
    size_t n = (size_t)BB * LL * KP;
    prep_patches<<<dim3((unsigned)((n + 255) / 256)), dim3(256), 0, stream>>>(x, P);
  }
  prep_scl_mm<<<dim3(BB * LL), dim3(256), 0, stream>>>(x, mask, scl, mm);
  prep_w<<<dim3((9 * C2 * CC + 255) / 256), dim3(256), 0, stream>>>(cw, c1w, c2w, hcw, hc1w, hc2w);
  {
    size_t n = (size_t)BB * LL * CC;
    prep_xcat<<<dim3((unsigned)((n + 255) / 256)), dim3(256), 0, stream>>>(x, ycat);
  }
  // attention pipeline
  gemm_scores<<<dim3(BB, LL / 64, LL / 128), dim3(256), 0, stream>>>(P, scl, S);
  fuse_softmax<<<dim3(BB * LL), dim3(256), 0, stream>>>(S, mm, Amat);
  gemm_deconv<<<dim3(BB, LL / 64, CC / 128), dim3(256), 0, stream>>>(Amat, P, ycat);
  // trailing gated convs
  gemm_out<<<dim3(BB, LL / 64, CC / 128), dim3(256), 0, stream>>>(ycat, hcw, hc1w, hc2w,
                                                                  cb, c1b, c2b, out);
  // second output: ones_like(mask)
  fill_ones<<<dim3((BB * LL + 255) / 256), dim3(256), 0, stream>>>(out + (size_t)BB * LL * CC);
}
